// AttentionBlock_58162447122671
// MI455X (gfx1250) — matrix operand order repaired and checked by reading, not compiled, not run on hardware
//
#include <hip/hip_runtime.h>
#include <hip/hip_bf16.h>
#include <math.h>

typedef _Float16 h16;
typedef __attribute__((ext_vector_type(16))) _Float16 v16h;
typedef __attribute__((ext_vector_type(8)))  _Float16 v8h;
typedef __attribute__((ext_vector_type(8)))  float    v8f;

union Frag { v16h v; v8h h[2]; };

static __device__ __forceinline__ v8f wmma32(const Frag& a, const Frag& b, v8f c) {
  // D = A(16x32 f16) * B(32x16 f16) + C(16x16 f32)
  return __builtin_amdgcn_wmma_f32_16x16x32_f16(false, a.v, false, b.v, (short)0, c, false, false);
}

#define BB   4
#define CC   256
#define NN   4096
#define NGRP 8
#define CPG  32          // channels per group
#define GELM (CPG * NN)  // elements per (b,group) = 131072

// ---------------------------------------------------------------- weights f32 -> f16
__global__ __launch_bounds__(256) void wconv(const float* __restrict__ wq, const float* __restrict__ wk,
                                             const float* __restrict__ wv, const float* __restrict__ wp,
                                             h16* __restrict__ wqh, h16* __restrict__ wkh,
                                             h16* __restrict__ wvh, h16* __restrict__ wph) {
  int i = blockIdx.x * 256 + threadIdx.x;   // grid = 256 blocks -> exactly 65536
  wqh[i] = (h16)wq[i];
  wkh[i] = (h16)wk[i];
  wvh[i] = (h16)wv[i];
  wph[i] = (h16)wp[i];
}

// ---------------------------------------------------------------- groupnorm stats
__global__ __launch_bounds__(256) void gn_stats(const float* __restrict__ x, float* __restrict__ stats) {
  __shared__ float ssum[256];
  __shared__ float ssq[256];
  int bg = blockIdx.x;                                   // b*8+g, 32 blocks
  const float* base = x + (size_t)bg * GELM;
  float s = 0.f, q = 0.f;
  for (int i = threadIdx.x; i < GELM; i += 256) { float v = base[i]; s += v; q += v * v; }
  ssum[threadIdx.x] = s; ssq[threadIdx.x] = q;
  __syncthreads();
  for (int st = 128; st > 0; st >>= 1) {
    if (threadIdx.x < st) { ssum[threadIdx.x] += ssum[threadIdx.x + st]; ssq[threadIdx.x] += ssq[threadIdx.x + st]; }
    __syncthreads();
  }
  if (threadIdx.x == 0) {
    float mean = ssum[0] * (1.f / GELM);
    float var  = ssq[0] * (1.f / GELM) - mean * mean;
    stats[bg * 2 + 0] = mean;
    stats[bg * 2 + 1] = rsqrtf(var + 1e-5f);
  }
}

// ---------------------------------------------------------------- normalize + transpose -> xnT[b][n][c] f16
__global__ __launch_bounds__(256) void gn_apply(const float* __restrict__ x, const float* __restrict__ gamma,
                                                const float* __restrict__ beta, const float* __restrict__ stats,
                                                h16* __restrict__ xnT) {
  __shared__ float tile[32][33];
  int b = blockIdx.z;
  int c0 = blockIdx.x * 32;   // group index == blockIdx.x (0..7)
  int n0 = blockIdx.y * 32;
  int tx = threadIdx.x, ty = threadIdx.y;  // (32,8)
  float mean = stats[(b * NGRP + blockIdx.x) * 2 + 0];
  float rstd = stats[(b * NGRP + blockIdx.x) * 2 + 1];
  #pragma unroll
  for (int j = 0; j < 32; j += 8) {
    int c = c0 + ty + j;
    float v = x[((size_t)b * CC + c) * NN + n0 + tx];
    tile[ty + j][tx] = (v - mean) * rstd * gamma[c] + beta[c];
  }
  __syncthreads();
  #pragma unroll
  for (int j = 0; j < 32; j += 8) {
    int n = n0 + ty + j;
    xnT[((size_t)b * NN + n) * CC + c0 + tx] = (h16)tile[tx][ty + j];
  }
}

// ---------------------------------------------------------------- 16x64-tile GEMM (A-fragment reuse x4):
// D[n,o] = sum_c A[n,c] * W[o,c]   (A: [b][n][c] f16, W: [o][c] f16)
// mode 0: outh[b][n][o] = (D+bias)*scale   (q/k; q folds softmax scale & log2e)
// mode 1: outh[b][o][n] = (D+bias)         (v, stored [c][n] for PV B-operand)
// mode 2: outf[b][o][n] = D+bias+resid     (projection + residual, f32)
__global__ __launch_bounds__(256) void gemm16(const h16* __restrict__ A, const h16* __restrict__ W,
                                              const float* __restrict__ bias, const float* __restrict__ resid,
                                              h16* __restrict__ outh, float* __restrict__ outf,
                                              int mode, float scale) {
  int wid  = (blockIdx.x * 256 + threadIdx.x) >> 5;   // 4096 waves
  int lane = threadIdx.x & 31;
  int og = wid & 3;            // group of 4 o-tiles (64 outputs)
  int nt = (wid >> 2) & 255;
  int b  = wid >> 10;
  int row16 = lane & 15, hi = lane >> 4;

  const h16* arow = A + ((size_t)b * NN + nt * 16 + row16) * CC + hi * 8;
  const h16* brow = W + (size_t)(og * 64 + row16) * CC + hi * 8;

  v8f acc[4];
  #pragma unroll
  for (int j = 0; j < 4; ++j) { v8f z = {}; acc[j] = z; }

  #pragma unroll
  for (int kk = 0; kk < 8; ++kk) {
    Frag a;
    a.h[0] = *(const v8h*)(arow + kk * 32);
    a.h[1] = *(const v8h*)(arow + kk * 32 + 16);
    #pragma unroll
    for (int j = 0; j < 4; ++j) {
      Frag w;
      const h16* bp = brow + (size_t)j * 16 * CC + kk * 32;
      w.h[0] = *(const v8h*)(bp);
      w.h[1] = *(const v8h*)(bp + 16);
      acc[j] = wmma32(a, w, acc[j]);
    }
  }

  #pragma unroll
  for (int j = 0; j < 4; ++j) {
    int o = og * 64 + j * 16 + row16;
    float bval = bias[o];
    #pragma unroll
    for (int r = 0; r < 8; ++r) {
      int n = nt * 16 + r + hi * 8;
      if (mode == 0) {
        outh[((size_t)b * NN + n) * CC + o] = (h16)((acc[j][r] + bval) * scale);
      } else if (mode == 1) {
        outh[((size_t)b * CC + o) * NN + n] = (h16)(acc[j][r] + bval);
      } else {
        size_t idx = ((size_t)b * CC + o) * NN + n;
        outf[idx] = acc[j][r] + bval + resid[idx];
      }
    }
  }
}

// ---------------------------------------------------------------- flash attention
// per wave: one (b, 16 query rows). qT/kT: [b][n][c] f16 (q pre-scaled by log2e/16 -> scores in
// log2 domain; softmax exactly invariant). V: [b][c][n] f16. oT out: [b][n][c] f16.
__global__ __launch_bounds__(128) void attn(const h16* __restrict__ qT, const h16* __restrict__ kT,
                                            const h16* __restrict__ V, h16* __restrict__ oT) {
  __shared__ __align__(16) h16 ps[4][16][32];   // per-wave P tile: C-layout -> A-layout transform
  int w    = threadIdx.x >> 5;
  int wid  = blockIdx.x * 4 + w;                // 1024 waves
  int b    = wid >> 8;
  int nt   = wid & 255;
  int lane = threadIdx.x & 31;
  int row16 = lane & 15, hi = lane >> 4;

  const h16* qrow  = qT + ((size_t)b * NN + nt * 16 + row16) * CC + hi * 8;
  const h16* kbase = kT + (size_t)b * NN * CC;
  const h16* vbase = V  + (size_t)b * CC * NN + (size_t)row16 * NN + hi * 8;

  v8f O[16];
  float mrun[8], lsum[8];
  #pragma unroll
  for (int ct = 0; ct < 16; ++ct) { v8f z = {}; O[ct] = z; }
  #pragma unroll
  for (int r = 0; r < 8; ++r) { mrun[r] = -3.0e38f; lsum[r] = 0.f; }

  for (int mc = 0; mc < 128; ++mc) {            // m scanned in chunks of 32
    int m0 = mc * 32;
    const h16* k0 = kbase + (size_t)(m0 + row16) * CC + hi * 8;
    const h16* k1 = k0 + (size_t)16 * CC;
    v8f S0 = {}, S1 = {};
    #pragma unroll
    for (int kk = 0; kk < 8; ++kk) {
      Frag a, b0, b1;
      a.h[0]  = *(const v8h*)(qrow + kk * 32);
      a.h[1]  = *(const v8h*)(qrow + kk * 32 + 16);
      b0.h[0] = *(const v8h*)(k0 + kk * 32);
      b0.h[1] = *(const v8h*)(k0 + kk * 32 + 16);
      b1.h[0] = *(const v8h*)(k1 + kk * 32);
      b1.h[1] = *(const v8h*)(k1 + kk * 32 + 16);
      S0 = wmma32(a, b0, S0);
      S1 = wmma32(a, b1, S1);
    }
    // online softmax on the 16x32 score chunk (row stats per half-wave, C-layout aligned);
    // scores are in log2 domain -> bare v_exp_f32 via exp2f
    #pragma unroll
    for (int r = 0; r < 8; ++r) {
      float t = fmaxf(S0[r], S1[r]);
      t = fmaxf(t, __shfl_xor(t, 1));
      t = fmaxf(t, __shfl_xor(t, 2));
      t = fmaxf(t, __shfl_xor(t, 4));
      t = fmaxf(t, __shfl_xor(t, 8));
      float nm = fmaxf(mrun[r], t);
      float al = exp2f(mrun[r] - nm);
      mrun[r] = nm;
      float p0 = exp2f(S0[r] - nm);
      float p1 = exp2f(S1[r] - nm);
      ps[w][r + hi * 8][row16]      = (h16)p0;
      ps[w][r + hi * 8][row16 + 16] = (h16)p1;
      float rs = p0 + p1;
      rs += __shfl_xor(rs, 1);
      rs += __shfl_xor(rs, 2);
      rs += __shfl_xor(rs, 4);
      rs += __shfl_xor(rs, 8);
      lsum[r] = lsum[r] * al + rs;
      #pragma unroll
      for (int ct = 0; ct < 16; ++ct) O[ct][r] *= al;
    }
    __builtin_amdgcn_wave_barrier();            // keep DS store->load order (in-order per wave)
    Frag pa;
    pa.h[0] = *(const v8h*)&ps[w][row16][hi * 8];
    pa.h[1] = *(const v8h*)&ps[w][row16][hi * 8 + 16];
    __builtin_amdgcn_wave_barrier();
    const h16* vr = vbase + m0;
    #pragma unroll
    for (int ct = 0; ct < 16; ++ct) {
      Frag vb;
      const h16* vp = vr + (size_t)ct * 16 * NN;
      vb.h[0] = *(const v8h*)(vp);
      vb.h[1] = *(const v8h*)(vp + 16);
      O[ct] = wmma32(pa, vb, O[ct]);
    }
  }

  float inv[8];
  #pragma unroll
  for (int r = 0; r < 8; ++r) inv[r] = 1.0f / lsum[r];
  #pragma unroll
  for (int ct = 0; ct < 16; ++ct) {
    #pragma unroll
    for (int r = 0; r < 8; ++r) {
      int n = nt * 16 + r + hi * 8;
      int c = ct * 16 + row16;
      oT[((size_t)b * NN + n) * CC + c] = (h16)(O[ct][r] * inv[r]);
    }
  }
}

// ---------------------------------------------------------------- launch
extern "C" void kernel_launch(void* const* d_in, const int* in_sizes, int n_in,
                              void* d_out, int out_size, void* d_ws, size_t ws_size,
                              hipStream_t stream) {
  (void)in_sizes; (void)n_in; (void)out_size; (void)ws_size;
  const float* x     = (const float*)d_in[0];
  const float* gamma = (const float*)d_in[1];
  const float* beta  = (const float*)d_in[2];
  const float* wq    = (const float*)d_in[3];
  const float* bq    = (const float*)d_in[4];
  const float* wk    = (const float*)d_in[5];
  const float* bk    = (const float*)d_in[6];
  const float* wv    = (const float*)d_in[7];
  const float* bv    = (const float*)d_in[8];
  const float* wp    = (const float*)d_in[9];
  const float* bp    = (const float*)d_in[10];
  float* out = (float*)d_out;

  char* ws = (char*)d_ws;
  const size_t TSZ = (size_t)BB * NN * CC * sizeof(h16);  // 8 MB per f16 tensor
  h16* xnT = (h16*)(ws);
  h16* qT  = (h16*)(ws + TSZ);
  h16* kT  = (h16*)(ws + 2 * TSZ);
  h16* vB  = (h16*)(ws + 3 * TSZ);
  h16* oT  = (h16*)(ws + 4 * TSZ);
  char* wsw = ws + 5 * TSZ;
  const size_t WSZ = (size_t)CC * CC * sizeof(h16);       // 128 KB
  h16* wqh = (h16*)(wsw);
  h16* wkh = (h16*)(wsw + WSZ);
  h16* wvh = (h16*)(wsw + 2 * WSZ);
  h16* wph = (h16*)(wsw + 3 * WSZ);
  float* stats = (float*)(wsw + 4 * WSZ);                 // 64 floats

  wconv<<<256, 256, 0, stream>>>(wq, wk, wv, wp, wqh, wkh, wvh, wph);
  gn_stats<<<BB * NGRP, 256, 0, stream>>>(x, stats);
  gn_apply<<<dim3(CC / 32, NN / 32, BB), dim3(32, 8), 0, stream>>>(x, gamma, beta, stats, xnT);

  // q scale = (1/sqrt(256)) * log2(e): scores land in log2 domain for bare v_exp_f32
  const float scl = 0.0625f * 1.44269504088896340736f;
  gemm16<<<512, 256, 0, stream>>>(xnT, wqh, bq, nullptr, qT, nullptr, 0, scl);
  gemm16<<<512, 256, 0, stream>>>(xnT, wkh, bk, nullptr, kT, nullptr, 0, 1.0f);
  gemm16<<<512, 256, 0, stream>>>(xnT, wvh, bv, nullptr, vB, nullptr, 1, 1.0f);

  attn<<<256, 128, 0, stream>>>(qT, kT, vB, oT);

  gemm16<<<512, 256, 0, stream>>>(oT, wph, bp, x, nullptr, out, 2, 1.0f);
}
